// kmeans_cluster_18459769439016
// MI455X (gfx1250) — matrix operand order repaired and checked by reading, not compiled, not run on hardware
//
#include <hip/hip_runtime.h>
#include <hip/hip_bf16.h>

// Problem constants (from reference: B=4096, D=1024, K=256)
#define NB 4096
#define ND 1024
#define NK 256

typedef __attribute__((ext_vector_type(16))) _Float16 v16h;
typedef __attribute__((ext_vector_type(8)))  _Float16 v8h;
typedef __attribute__((ext_vector_type(4)))  _Float16 h4;
typedef __attribute__((ext_vector_type(8)))  float    v8f;

// -------------------------------------------------------------------------
// K1: L2-normalize rows of X[R x 1024] (f32) -> Y (f16). One block per row.
// -------------------------------------------------------------------------
__global__ void normalize_rows_f16(const float* __restrict__ X,
                                   _Float16* __restrict__ Y) {
    __shared__ float red[256];
    const int t = threadIdx.x;
    const size_t row = blockIdx.x;
    const float4 x = *(const float4*)(X + row * ND + t * 4);
    red[t] = x.x * x.x + x.y * x.y + x.z * x.z + x.w * x.w;
    __syncthreads();
    for (int s = 128; s > 0; s >>= 1) {
        if (t < s) red[t] += red[t + s];
        __syncthreads();
    }
    const float inv = 1.0f / fmaxf(sqrtf(red[0]), 1e-8f);
    h4 y;
    y.x = (_Float16)(x.x * inv);
    y.y = (_Float16)(x.y * inv);
    y.z = (_Float16)(x.z * inv);
    y.w = (_Float16)(x.w * inv);
    *(h4*)(Y + row * ND + t * 4) = y;
}

// -------------------------------------------------------------------------
// K2: intra_cos_sim[4096x256] = dn (4096x1024 f16) x cn^T (f16), f32 accum.
// 512 threads = 16 waves per block; wave w computes the 16x16 tile at
// (mbase = blockIdx.x*16, nbase = w*16). K-loop steps by 32 per WMMA.
// -------------------------------------------------------------------------
__global__ void __launch_bounds__(512)
gemm_cos_wmma(const _Float16* __restrict__ dn,
              const _Float16* __restrict__ cn,
              float* __restrict__ sim) {
    const int wave = threadIdx.x >> 5;
    const int lane = threadIdx.x & 31;
    const int hl   = lane >> 4;   // lane half: 0 or 1
    const int r    = lane & 15;

    const int mbase = blockIdx.x * 16;
    const int nbase = wave * 16;

    // A: lane holds dn row (mbase + r). B: lane holds cn row (nbase + r).
    const _Float16* pA = dn + (size_t)(mbase + r) * ND;
    const _Float16* pB = cn + (size_t)(nbase + r) * ND;

    v8f acc = {};
    for (int k0 = 0; k0 < ND; k0 += 32) {
        // A fragment (16-bit A 16x32 layout): VGPRs 0-3 <- K = k0+8*hl+[0..8),
        // VGPRs 4-7 <- K = k0+16+8*hl+[0..8)
        v8h a0 = *(const v8h*)(pA + k0 + 8 * hl);
        v8h a1 = *(const v8h*)(pA + k0 + 16 + 8 * hl);
        // B fragment (32x16): lane holds column n, K = k0+8*hl+[0..8), then k0+16+8*hl+[0..8)
        v8h b0 = *(const v8h*)(pB + k0 + 8 * hl);
        v8h b1 = *(const v8h*)(pB + k0 + 16 + 8 * hl);

        v16h A = __builtin_shufflevector(a0, a1, 0, 1, 2, 3, 4, 5, 6, 7,
                                                 8, 9, 10, 11, 12, 13, 14, 15);
        v16h B = __builtin_shufflevector(b0, b1, 0, 1, 2, 3, 4, 5, 6, 7,
                                                 8, 9, 10, 11, 12, 13, 14, 15);
        acc = __builtin_amdgcn_wmma_f32_16x16x32_f16(
            /*neg_a=*/false, A, /*neg_b=*/false, B,
            /*c_mod=*/(short)0, acc, /*reuse_a=*/false, /*reuse_b=*/false);
    }

    // D layout: acc[j] -> (m = mbase + j + 8*hl, n = nbase + r)
    float* po = sim + (size_t)mbase * NK + nbase;
#pragma unroll
    for (int j = 0; j < 8; ++j)
        po[(size_t)(j + 8 * hl) * NK + r] = acc[j];
}

// -------------------------------------------------------------------------
// K3: per-row top-2 over 256 sims + gathers. One block (256 thr) per row.
// Tie rule: lower index wins (matches jnp.argmax / top_k ordering).
// -------------------------------------------------------------------------
__global__ void top2_gather(const float* __restrict__ sim,
                            const float* __restrict__ centroid,
                            float* __restrict__ o_dpidx,
                            float* __restrict__ o_hnidx,
                            float* __restrict__ o_top2,
                            float* __restrict__ o_dpcent,
                            float* __restrict__ o_hardneg,
                            int* __restrict__ ws_idx) {
    __shared__ float sV[256];
    __shared__ int   sI[256];
    __shared__ int   sTop[2];
    __shared__ float sVal2;

    const int t = threadIdx.x;
    const size_t row = blockIdx.x;
    const float v = sim[row * NK + t];

    // pass 1: argmax
    sV[t] = v; sI[t] = t;
    __syncthreads();
    for (int s = 128; s > 0; s >>= 1) {
        if (t < s) {
            float v2 = sV[t + s]; int i2 = sI[t + s];
            if (v2 > sV[t] || (v2 == sV[t] && i2 < sI[t])) { sV[t] = v2; sI[t] = i2; }
        }
        __syncthreads();
    }
    if (t == 0) sTop[0] = sI[0];
    __syncthreads();
    const int idx1 = sTop[0];

    // pass 2: second-largest (exclude idx1)
    sV[t] = (t == idx1) ? -1e30f : v;
    sI[t] = t;
    __syncthreads();
    for (int s = 128; s > 0; s >>= 1) {
        if (t < s) {
            float v2 = sV[t + s]; int i2 = sI[t + s];
            if (v2 > sV[t] || (v2 == sV[t] && i2 < sI[t])) { sV[t] = v2; sI[t] = i2; }
        }
        __syncthreads();
    }
    if (t == 0) { sTop[1] = sI[0]; sVal2 = sV[0]; }
    __syncthreads();
    const int idx2 = sTop[1];

    if (t == 0) {
        o_dpidx[row] = (float)idx1;
        o_hnidx[row] = (float)idx2;
        o_top2[row]  = sVal2;
        ws_idx[row]  = idx1;
    }

    // gather original (un-normalized) centroid rows: 256 thr x float4 = 1024 f32
    const float4* c1 = (const float4*)(centroid + (size_t)idx1 * ND);
    const float4* c2 = (const float4*)(centroid + (size_t)idx2 * ND);
    ((float4*)(o_dpcent  + row * ND))[t] = c1[t];
    ((float4*)(o_hardneg + row * ND))[t] = c2[t];
}

// -------------------------------------------------------------------------
// K4: dp_cluster[i][j] = (idx[i]==idx[j] && i!=j). One block (1024 thr) per
// row; each thread writes a float4.
// -------------------------------------------------------------------------
__global__ void __launch_bounds__(1024)
cluster_matrix(const int* __restrict__ idx, float* __restrict__ o_cluster) {
    const int i = blockIdx.x;
    const int ri = idx[i];
    const int j0 = threadIdx.x * 4;
    const int4 ij = *(const int4*)(idx + j0);
    float4 o;
    o.x = (ij.x == ri && (j0 + 0) != i) ? 1.0f : 0.0f;
    o.y = (ij.y == ri && (j0 + 1) != i) ? 1.0f : 0.0f;
    o.z = (ij.z == ri && (j0 + 2) != i) ? 1.0f : 0.0f;
    o.w = (ij.w == ri && (j0 + 3) != i) ? 1.0f : 0.0f;
    *(float4*)(o_cluster + (size_t)i * NB + j0) = o;
}

// -------------------------------------------------------------------------
// K5: index_dp[k][b] = (idx[b]==k), stored as float. float4 per thread.
// -------------------------------------------------------------------------
__global__ void index_dp_matrix(const int* __restrict__ idx,
                                float* __restrict__ o_indexdp) {
    const int g  = blockIdx.x * blockDim.x + threadIdx.x;  // group of 4 b's
    const int e0 = g * 4;
    const int b0 = e0 & (NB - 1);
    const int k  = e0 >> 12;  // / 4096
    const int4 ib = *(const int4*)(idx + b0);
    float4 o;
    o.x = (ib.x == k) ? 1.0f : 0.0f;
    o.y = (ib.y == k) ? 1.0f : 0.0f;
    o.z = (ib.z == k) ? 1.0f : 0.0f;
    o.w = (ib.w == k) ? 1.0f : 0.0f;
    *(float4*)(o_indexdp + (size_t)e0) = o;
}

// -------------------------------------------------------------------------
extern "C" void kernel_launch(void* const* d_in, const int* in_sizes, int n_in,
                              void* d_out, int out_size, void* d_ws, size_t ws_size,
                              hipStream_t stream) {
    const float* datapoints = (const float*)d_in[0];   // 4096 x 1024
    const float* centroid   = (const float*)d_in[1];   // 256 x 1024
    // d_in[2] (batch_cos_sim) is unused by the reference outputs.

    // Output layout (flat f32, reference tuple order)
    float* o_sim     = (float*)d_out;                          // 4096*256
    float* o_dpidx   = o_sim     + (size_t)NB * NK;            // 4096
    float* o_cluster = o_dpidx   + NB;                         // 4096*4096
    float* o_indexdp = o_cluster + (size_t)NB * NB;            // 256*4096
    float* o_dpcent  = o_indexdp + (size_t)NK * NB;            // 4096*1024
    float* o_hardneg = o_dpcent  + (size_t)NB * ND;            // 4096*1024
    float* o_hnidx   = o_hardneg + (size_t)NB * ND;            // 4096
    float* o_top2    = o_hnidx   + NB;                         // 4096

    // Workspace: dn f16 (8MB) | cn f16 (0.5MB) | argmax ints (16KB)
    _Float16* dn_h = (_Float16*)d_ws;
    _Float16* cn_h = dn_h + (size_t)NB * ND;
    int* ws_idx    = (int*)(cn_h + (size_t)NK * ND);

    normalize_rows_f16<<<NB, 256, 0, stream>>>(datapoints, dn_h);
    normalize_rows_f16<<<NK, 256, 0, stream>>>(centroid, cn_h);

    gemm_cos_wmma<<<NB / 16, 512, 0, stream>>>(dn_h, cn_h, o_sim);

    top2_gather<<<NB, 256, 0, stream>>>(o_sim, centroid, o_dpidx, o_hnidx,
                                        o_top2, o_dpcent, o_hardneg, ws_idx);

    cluster_matrix<<<NB, 1024, 0, stream>>>(ws_idx, o_cluster);

    index_dp_matrix<<<(NK * NB / 4) / 256, 256, 0, stream>>>(ws_idx, o_indexdp);
}
